// EntityModel_14345190768675
// MI455X (gfx1250) — matrix operand order repaired and checked by reading, not compiled, not run on hardware
//
#include <hip/hip_runtime.h>
#include <math.h>

typedef __attribute__((ext_vector_type(16))) __bf16 v16bf;
typedef __attribute__((ext_vector_type(8)))  __bf16 v8bf;
typedef __attribute__((ext_vector_type(8)))  float  v8f;

#define HDIM  768
#define TAGS  9
#define BATCH 64
#define SEQ   512
#define NROWS (BATCH * SEQ)   // 32768
#define KT    (HDIM / 32)     // 24 K-tiles

// ---------------- workspace layout (bytes) ----------------
// [0)              W_fclT  bf16 [768][768]   1,179,648
// [1179648)        W_tagT  bf16 [16][768]       24,576
// [1204224)        emissions f32 [32768][9]  1,179,648
// [2383872)        llh f32 [64]                    256
#define WS_WFCLT 0
#define WS_WTAGT 1179648
#define WS_EM    1204224
#define WS_LLH   2383872

// -------- prep: W_fclT[n][k] = bf16(W_fcl[k][n]) --------
__global__ void k_prep_wfclT(const float* __restrict__ w, unsigned short* __restrict__ wt_raw) {
    __bf16* wt = (__bf16*)wt_raw;
    int i = blockIdx.x * 256 + threadIdx.x;      // over 768*768
    int n = i / HDIM, k = i % HDIM;
    wt[i] = (__bf16)w[k * HDIM + n];
}

// -------- prep: W_tagT[n][k], n padded to 16 with zeros --------
__global__ void k_prep_wtagT(const float* __restrict__ w, unsigned short* __restrict__ wt_raw) {
    __bf16* wt = (__bf16*)wt_raw;
    int i = blockIdx.x * 256 + threadIdx.x;      // over 16*768
    int n = i / HDIM, k = i % HDIM;
    float v = (n < TAGS) ? w[k * TAGS + n] : 0.f;
    wt[i] = (__bf16)v;
}

// -------- fused head: emissions = tanh(hidden@Wf + bf) @ Wt + bt --------
// Block = 512 threads (16 waves), M-tile = 32 rows, N = full 768.
// Wave grid: wm in {0,1} (M half), wn in 0..7 (96-col group, 6 WMMA tiles).
__global__ __launch_bounds__(512)
void k_head(const float* __restrict__ hidden,
            const unsigned short* __restrict__ WfT_raw,
            const float* __restrict__ bfcl,
            const unsigned short* __restrict__ WtT_raw,
            const float* __restrict__ btag,
            float* __restrict__ em) {
    const __bf16* WfT = (const __bf16*)WfT_raw;
    const __bf16* WtT = (const __bf16*)WtT_raw;

    __shared__ __align__(16) __bf16 Asm[2][32][40];   // double-buffered 32x32 A tile
    __shared__ __align__(16) __bf16 Hsm[32][776];     // 32 x 768 h tile (padded)
    __shared__ float emAcc[32][16];                   // stage-2 partial sums

    const int tid  = threadIdx.x;
    const int lane = tid & 31;
    const int w    = tid >> 5;              // 16 waves
    const int wm   = w >> 3;                // M half (0/1)
    const int wn   = w & 7;                 // N group (0..7)
    const int r0   = blockIdx.x * 32;       // global row base
    const int nl   = lane & 15;
    const int khalf = (lane < 16) ? 0 : 8;  // A-fragment K base (ISA 16-bit A layout)
    const int kq    = (lane >> 4) << 3;     // B-fragment K base (ISA 16-bit B layout)
    const int mofs  = (lane < 16) ? 0 : 8;  // C/D row offset per lane half

    // staging assignment: 1024 f32 by 512 threads, 2 each
    const int ar = (tid * 2) >> 5;          // 0..31
    const int ac = (tid * 2) & 31;          // even

    v8f zero = {};
    v8f acc[6];
    #pragma unroll
    for (int i = 0; i < 6; ++i) acc[i] = zero;

    // prologue: stage K-tile 0, zero stage-2 accumulator
    {
        const float* gp = hidden + (size_t)(r0 + ar) * HDIM + ac;
        Asm[0][ar][ac]     = (__bf16)gp[0];
        Asm[0][ar][ac + 1] = (__bf16)gp[1];
        ((float*)emAcc)[tid] = 0.f;
    }
    __syncthreads();

    for (int kt = 0; kt < KT; ++kt) {
        const int cur = kt & 1;
        const int k0 = kt * 32;
        const bool more = (kt + 1) < KT;

        // A fragment from LDS
        v16bf a;
        {
            const __bf16* base = &Asm[cur][wm * 16 + nl][khalf];
            v8bf lo = *(const v8bf*)base;
            v8bf hi = *(const v8bf*)(base + 16);
            #pragma unroll
            for (int i = 0; i < 8; ++i) { a[i] = lo[i]; a[i + 8] = hi[i]; }
        }

        // prefetch next A tile into registers (wait lands after the WMMA block)
        float f0n = 0.f, f1n = 0.f;
        if (more) {
            const float* gp = hidden + (size_t)(r0 + ar) * HDIM + (k0 + 32) + ac;
            f0n = gp[0];
            f1n = gp[1];
            if (tid < 32 && (kt + 2) < KT)
                __builtin_prefetch(hidden + (size_t)(r0 + tid) * HDIM + k0 + 64, 0, 0);
        }

        #pragma unroll
        for (int i = 0; i < 6; ++i) {
            const int n = wn * 96 + i * 16 + nl;
            const __bf16* bp = WfT + (size_t)n * HDIM + k0 + kq;
            v8bf blo = *(const v8bf*)bp;
            v8bf bhi = *(const v8bf*)(bp + 16);
            v16bf b;
            #pragma unroll
            for (int j = 0; j < 8; ++j) { b[j] = blo[j]; b[j + 8] = bhi[j]; }
            acc[i] = __builtin_amdgcn_wmma_f32_16x16x32_bf16(
                false, a, false, b, (short)0, acc[i], false, false);
        }

        if (more) {
            Asm[cur ^ 1][ar][ac]     = (__bf16)f0n;
            Asm[cur ^ 1][ar][ac + 1] = (__bf16)f1n;
        }
        __syncthreads();
    }

    // epilogue: bias + tanh -> bf16 h tile in LDS
    #pragma unroll
    for (int i = 0; i < 6; ++i) {
        const int gcol = wn * 96 + i * 16 + nl;
        const float bias = bfcl[gcol];
        #pragma unroll
        for (int e = 0; e < 8; ++e) {
            Hsm[wm * 16 + e + mofs][gcol] = (__bf16)tanhf(acc[i][e] + bias);
        }
    }
    __syncthreads();

    // stage 2: emissions[32][9] = h @ W_tagT, split 3 K-tiles per wave,
    // partial C combined through LDS float atomics.
    {
        v8f c = zero;
        const int arow = wm * 16 + nl;
        #pragma unroll
        for (int q = 0; q < 3; ++q) {
            const int kb = (wn * 3 + q) * 32;
            v16bf a2, b2;
            {
                const __bf16* base = &Hsm[arow][kb + khalf];
                v8bf lo = *(const v8bf*)base;
                v8bf hi = *(const v8bf*)(base + 16);
                #pragma unroll
                for (int i = 0; i < 8; ++i) { a2[i] = lo[i]; a2[i + 8] = hi[i]; }
            }
            const __bf16* bp = WtT + (size_t)nl * HDIM + kb + kq;
            v8bf blo = *(const v8bf*)bp;
            v8bf bhi = *(const v8bf*)(bp + 16);
            #pragma unroll
            for (int j = 0; j < 8; ++j) { b2[j] = blo[j]; b2[j + 8] = bhi[j]; }
            c = __builtin_amdgcn_wmma_f32_16x16x32_bf16(
                false, a2, false, b2, (short)0, c, false, false);
        }
        #pragma unroll
        for (int e = 0; e < 8; ++e)
            atomicAdd(&emAcc[wm * 16 + e + mofs][nl], c[e]);
    }
    __syncthreads();

    {
        const int m = tid >> 4;      // 0..31
        const int n = tid & 15;
        if (n < TAGS)
            em[(size_t)(r0 + m) * TAGS + n] = emAcc[m][n] + btag[n];
    }
}

// -------- CRF: llh numerator + forward logsumexp + Viterbi, one wave per batch --------
__global__ __launch_bounds__(32)
void k_crf(const float* __restrict__ em, const int* __restrict__ mask,
           const int* __restrict__ tags, const float* __restrict__ start,
           const float* __restrict__ endt, const float* __restrict__ trans,
           float* __restrict__ llh, float* __restrict__ outTags) {
    __shared__ float transS[TAGS * TAGS];
    __shared__ float startS[TAGS], endS[TAGS];
    __shared__ unsigned char bp[SEQ][16];

    const int lane = threadIdx.x;
    const int b = blockIdx.x;
    for (int i = lane; i < TAGS * TAGS; i += 32) transS[i] = trans[i];
    if (lane < TAGS) { startS[lane] = start[lane]; endS[lane] = endt[lane]; }
    __syncthreads();

    const int jl = (lane < TAGS) ? lane : 0;
    float t[TAGS];
    #pragma unroll
    for (int j = 0; j < TAGS; ++j) t[j] = transS[j * TAGS + jl];   // trans[j][lane]

    const float NEG = -1e30f;
    const size_t base = (size_t)b * SEQ;
    float alpha = (lane < TAGS) ? (startS[lane] + em[base * TAGS + jl]) : NEG;
    float sv = alpha;
    int tag0 = tags[base];
    float score = startS[tag0] + em[base * TAGS + tag0];
    int prev = tag0;

    // software-pipelined loads for step s=1
    float e_c, et_c; int m_c, tg_c;
    {
        const size_t idx = base + 1;
        e_c  = em[idx * TAGS + jl];
        m_c  = mask[idx];
        tg_c = tags[idx];
        et_c = em[idx * TAGS + tg_c];
    }

    for (int s = 1; s < SEQ; ++s) {
        float e_n = 0.f, et_n = 0.f; int m_n = 0, tg_n = 0;
        if (s + 1 < SEQ) {   // prefetch next step
            const size_t nidx = base + s + 1;
            e_n  = em[nidx * TAGS + jl];
            m_n  = mask[nidx];
            tg_n = tags[nidx];
            et_n = em[nidx * TAGS + tg_n];
        }

        const float mf = (float)m_c;
        // numerator (uniform across lanes)
        score += mf * (transS[prev * TAGS + tg_c] + et_c);
        prev = (mf > 0.f) ? tg_c : prev;

        // forward (logsumexp) + viterbi (max/argmax) over previous tag j
        float va[TAGS], mA = NEG, mV = NEG;
        int bj = 0;
        #pragma unroll
        for (int j = 0; j < TAGS; ++j) {
            va[j] = __shfl(alpha, j, 32) + t[j];
            mA = fmaxf(mA, va[j]);
            float vs = __shfl(sv, j, 32) + t[j];
            if (vs > mV) { mV = vs; bj = j; }
        }
        float ssum = 0.f;
        #pragma unroll
        for (int j = 0; j < TAGS; ++j) ssum += __expf(va[j] - mA);
        const float nA = mA + __logf(ssum) + e_c;
        const float nV = mV + e_c;
        const bool on = mf > 0.f;
        alpha = on ? nA : alpha;
        sv    = on ? nV : sv;
        if (lane < 16) bp[s][lane] = (unsigned char)(on ? bj : lane);  // identity when masked

        e_c = e_n; m_c = m_n; tg_c = tg_n; et_c = et_n;
    }

    // denominator + viterbi terminal
    const float avd = (lane < TAGS) ? (alpha + endS[lane]) : NEG;
    const float svd = (lane < TAGS) ? (sv + endS[lane]) : NEG;
    float mD = NEG, mL = NEG; int last = 0;
    #pragma unroll
    for (int j = 0; j < TAGS; ++j) {
        mD = fmaxf(mD, __shfl(avd, j, 32));
        float y = __shfl(svd, j, 32);
        if (y > mL) { mL = y; last = j; }
    }
    float sD = 0.f;
    #pragma unroll
    for (int j = 0; j < TAGS; ++j) sD += __expf(__shfl(avd, j, 32) - mD);
    const float den = mD + __logf(sD);
    const float num = score + endS[prev];

    __syncthreads();
    if (lane == 0) {
        llh[b] = num - den;
        int tg = last;
        for (int s = SEQ - 1; s >= 1; --s) {
            outTags[base + s] = (float)tg;
            tg = bp[s][tg];
        }
        outTags[base] = (float)tg;
    }
}

// -------- loss = -mean(llh), one wave --------
__global__ __launch_bounds__(32)
void k_loss(const float* __restrict__ llh, float* __restrict__ out) {
    const int l = threadIdx.x;
    float s = llh[l] + llh[l + 32];
    #pragma unroll
    for (int o = 16; o > 0; o >>= 1) s += __shfl_xor(s, o, 32);
    if (l == 0) out[0] = -s / (float)BATCH;
}

extern "C" void kernel_launch(void* const* d_in, const int* in_sizes, int n_in,
                              void* d_out, int out_size, void* d_ws, size_t ws_size,
                              hipStream_t stream) {
    const float* hidden = (const float*)d_in[0];   // [64,512,768]
    const int*   mask   = (const int*)  d_in[1];   // [64,512]
    const int*   tags   = (const int*)  d_in[2];   // [64,512]
    const float* Wf     = (const float*)d_in[3];   // [768,768]
    const float* bf     = (const float*)d_in[4];   // [768]
    const float* Wt     = (const float*)d_in[5];   // [768,9]
    const float* bt     = (const float*)d_in[6];   // [9]
    const float* st     = (const float*)d_in[7];   // [9]
    const float* en     = (const float*)d_in[8];   // [9]
    const float* tr     = (const float*)d_in[9];   // [9,9]

    float* out = (float*)d_out;                    // [0]=loss, [1..]=tags
    char*  ws  = (char*)d_ws;
    unsigned short* WfT = (unsigned short*)(ws + WS_WFCLT);
    unsigned short* WtT = (unsigned short*)(ws + WS_WTAGT);
    float* em  = (float*)(ws + WS_EM);
    float* llh = (float*)(ws + WS_LLH);

    k_prep_wfclT<<<(HDIM * HDIM) / 256, 256, 0, stream>>>(Wf, WfT);
    k_prep_wtagT<<<(16 * HDIM) / 256,   256, 0, stream>>>(Wt, WtT);
    k_head<<<NROWS / 32, 512, 0, stream>>>(hidden, WfT, bf, WtT, bt, em);
    k_crf<<<BATCH, 32, 0, stream>>>(em, mask, tags, st, en, tr, llh, out + 1);
    k_loss<<<1, 32, 0, stream>>>(llh, out);
}
